// GatedGraphConvMLP_86062554677864
// MI455X (gfx1250) — matrix operand order repaired and checked by reading, not compiled, not run on hardware
//
#include <hip/hip_runtime.h>
#include <hip/hip_bf16.h>
#include <math.h>

#define NN 81920        // nodes
#define EE 1310720      // edges
#define HC 64
#define BS 512
#define KHEAD 10240     // G*EL*HC
#define SPLITK 10       // head GEMM K chunks of 1024

typedef __attribute__((ext_vector_type(16))) __bf16 v16bf;
typedef __attribute__((ext_vector_type(8)))  float  v8f;

__device__ __forceinline__ v8f wmma_bf16(v16bf a, v16bf b, v8f c) {
    return __builtin_amdgcn_wmma_f32_16x16x32_bf16(
        false, a, false, b, (short)0, c, false, false);
}

// A-operand pack: contiguous k order, same as the B packs (lane<16: K {kk+0..15};
// lane>=16: K {kk+16..31}); Arow points at A[row*lda].
__device__ __forceinline__ v16bf pack_a(const float* __restrict__ Arow, int kk, int half) {
    v16bf a;
    const float* p0 = Arow + kk + half * 16;
    const float* p1 = p0 + 8;
#pragma unroll
    for (int i = 0; i < 8; ++i) a[i] = (__bf16)p0[i];
#pragma unroll
    for (int i = 0; i < 8; ++i) a[8 + i] = (__bf16)p1[i];
    return a;
}

// B-operand where per-lane column data is contiguous in k (torch Linear [out,in]):
// B(k, j) = Wcol[k], element i -> k = kk + half*16 + i
__device__ __forceinline__ v16bf pack_b_contig(const float* __restrict__ Wcol, int kk, int half) {
    v16bf b;
    const float* p = Wcol + kk + half * 16;
#pragma unroll
    for (int i = 0; i < 16; ++i) b[i] = (__bf16)p[i];
    return b;
}

// B-operand for row-major K x N weight (Wg): B(k, j) = B[k*ldb + col]
__device__ __forceinline__ v16bf pack_b_rowmajor(const float* __restrict__ B, int ldb,
                                                 int col, int kk, int half) {
    v16bf b;
    const float* p = B + (size_t)(kk + half * 16) * ldb + col;
#pragma unroll
    for (int i = 0; i < 16; ++i) b[i] = (__bf16)p[i * ldb];
    return b;
}

__device__ __forceinline__ float sigmoidf_(float x) { return 1.0f / (1.0f + __expf(-x)); }

// ---------------- kernels ----------------

__global__ void pad_kernel(const float* __restrict__ x, float* __restrict__ h) {
    int idx = blockIdx.x * blockDim.x + threadIdx.x;   // N*64 threads
    int n = idx >> 6, f = idx & 63;
    h[idx] = (f < 5) ? x[n * 5 + f] : 0.0f;
}

// m[N,64] = h[N,64] @ Wg[64,64]  (Wg row-major K x N)
__global__ void gemm_m_kernel(const float* __restrict__ h, const float* __restrict__ Wg,
                              float* __restrict__ m) {
    int wave = (blockIdx.x * blockDim.x + threadIdx.x) >> 5;
    int lane = threadIdx.x & 31;
    if (wave >= (NN / 16) * 4) return;       // wave-uniform guard
    int rt = wave >> 2, ct = wave & 3;
    int r0 = rt * 16, c0 = ct * 16;
    int half = lane >> 4, lrow = lane & 15;
    const float* Arow = h + (size_t)(r0 + lrow) * HC;
    v8f acc = {0.f, 0.f, 0.f, 0.f, 0.f, 0.f, 0.f, 0.f};
#pragma unroll
    for (int kk = 0; kk < HC; kk += 32) {
        v16bf a = pack_a(Arow, kk, half);
        v16bf b = pack_b_rowmajor(Wg, HC, c0 + lrow, kk, half);
        acc = wmma_bf16(a, b, acc);
    }
    int col = c0 + lrow;
    int rbase = r0 + half * 8;
#pragma unroll
    for (int v = 0; v < 8; ++v)
        m[(size_t)(rbase + v) * HC + col] = acc[v];
}

// edge scatter: aggr[dst] += m[src] * w   (16 lanes per edge, 4 feats each)
__global__ void scatter_kernel(const int* __restrict__ ei, const float* __restrict__ ea,
                               const float* __restrict__ m, float* __restrict__ aggr) {
    long long idx = (long long)blockIdx.x * blockDim.x + threadIdx.x;  // E*16
    int e = (int)(idx >> 4);
    int c = (int)(idx & 15);
    int src = ei[e];
    int dst = ei[EE + e];
    float w = ea[e];
    const float4 mv = *reinterpret_cast<const float4*>(m + (size_t)src * HC + c * 4);
    float* b = aggr + (size_t)dst * HC + c * 4;
    atomicAdd(b + 0, mv.x * w);
    atomicAdd(b + 1, mv.y * w);
    atomicAdd(b + 2, mv.z * w);
    atomicAdd(b + 3, mv.w * w);
}

// fused GRU cell: block = 4 waves covering one 16-row strip, updates h in place.
__global__ void gru_kernel(float* __restrict__ h, const float* __restrict__ aggr,
                           const float* __restrict__ wih, const float* __restrict__ whh,
                           const float* __restrict__ bih, const float* __restrict__ bhh) {
    int r0 = blockIdx.x * 16;
    int w = threadIdx.x >> 5;        // col-tile 0..3
    int lane = threadIdx.x & 31;
    int c0 = w * 16;
    int half = lane >> 4, lrow = lane & 15;
    int cj = c0 + lrow;

    const float* Am = aggr + (size_t)(r0 + lrow) * HC;
    const float* Ah = h + (size_t)(r0 + lrow) * HC;
    const float* Wr = wih + (size_t)(0 * HC + cj) * HC;
    const float* Wz = wih + (size_t)(1 * HC + cj) * HC;
    const float* Wn = wih + (size_t)(2 * HC + cj) * HC;
    const float* Ur = whh + (size_t)(0 * HC + cj) * HC;
    const float* Uz = whh + (size_t)(1 * HC + cj) * HC;
    const float* Un = whh + (size_t)(2 * HC + cj) * HC;

    v8f ir = {0.f,0.f,0.f,0.f,0.f,0.f,0.f,0.f};
    v8f iz = ir, inn = ir, hr = ir, hz = ir, hn = ir;
#pragma unroll
    for (int kk = 0; kk < HC; kk += 32) {
        v16bf am = pack_a(Am, kk, half);
        ir  = wmma_bf16(am, pack_b_contig(Wr, kk, half), ir);
        iz  = wmma_bf16(am, pack_b_contig(Wz, kk, half), iz);
        inn = wmma_bf16(am, pack_b_contig(Wn, kk, half), inn);
        v16bf ah = pack_a(Ah, kk, half);
        hr  = wmma_bf16(ah, pack_b_contig(Ur, kk, half), hr);
        hz  = wmma_bf16(ah, pack_b_contig(Uz, kk, half), hz);
        hn  = wmma_bf16(ah, pack_b_contig(Un, kk, half), hn);
    }
    int rbase = r0 + half * 8;
    float hold[8];
#pragma unroll
    for (int v = 0; v < 8; ++v)
        hold[v] = h[(size_t)(rbase + v) * HC + cj];
    float bri = bih[cj],          brh = bhh[cj];
    float bzi = bih[HC + cj],     bzh = bhh[HC + cj];
    float bni = bih[2 * HC + cj], bnh = bhh[2 * HC + cj];

    __syncthreads();   // all reads of h done (block owns these 16 rows exclusively)

#pragma unroll
    for (int v = 0; v < 8; ++v) {
        float r = sigmoidf_((ir[v] + bri) + (hr[v] + brh));
        float z = sigmoidf_((iz[v] + bzi) + (hz[v] + bzh));
        float n = tanhf((inn[v] + bni) + r * (hn[v] + bnh));
        h[(size_t)(rbase + v) * HC + cj] = (1.0f - z) * n + z * hold[v];
    }
}

// z1[512,64] += hbig[512,10240] @ W1^T  with split-K and f32 atomics.
// The (bs g e)f -> bs (e g f) rearrange is folded into the A address math.
__global__ void head1_kernel(const float* __restrict__ h, const float* __restrict__ W1,
                             float* __restrict__ z1) {
    int gw = (blockIdx.x * blockDim.x + threadIdx.x) >> 5;
    int lane = threadIdx.x & 31;
    if (gw >= 32 * 4 * SPLITK) return;
    int rt = gw / (4 * SPLITK);
    int rem = gw % (4 * SPLITK);
    int ct = rem / SPLITK;
    int kc = rem % SPLITK;
    int r0 = rt * 16, c0 = ct * 16;
    int half = lane >> 4, lrow = lane & 15;
    int bs = r0 + lrow;
    int cj = c0 + lrow;
    const float* Wcol = W1 + (size_t)cj * KHEAD;
    v8f acc = {0.f, 0.f, 0.f, 0.f, 0.f, 0.f, 0.f, 0.f};
    int k0 = kc * (KHEAD / SPLITK);
    for (int kk = k0; kk < k0 + KHEAD / SPLITK; kk += 32) {
        // column index in rearranged matrix: k = (e*5+g)*64 + f
        int e = kk / 320;
        int g = (kk >> 6) % 5;
        int f0 = kk & 63;      // 0 or 32; a 32-wide k-step stays in one node row
        const float* Arow = h + ((size_t)((bs * 5 + g) * 32 + e)) * HC + f0;
        v16bf a = pack_a(Arow, 0, half);
        v16bf b = pack_b_contig(Wcol, kk, half);
        acc = wmma_bf16(a, b, acc);
    }
    int rbase = r0 + half * 8;
#pragma unroll
    for (int v = 0; v < 8; ++v)
        atomicAdd(&z1[(size_t)(rbase + v) * HC + cj], acc[v]);
}

__global__ void bias_relu_kernel(float* __restrict__ z1, const float* __restrict__ b1) {
    int idx = blockIdx.x * blockDim.x + threadIdx.x;   // 512*64
    float v = z1[idx] + b1[idx & 63];
    z1[idx] = fmaxf(v, 0.0f);
}

// tiny MLP tail + softmax: one thread per batch row
__global__ void tail_kernel(const float* __restrict__ z1,
                            const float* __restrict__ W2, const float* __restrict__ b2,
                            const float* __restrict__ W3, const float* __restrict__ b3,
                            const float* __restrict__ W4, const float* __restrict__ b4,
                            float* __restrict__ out) {
    int r = blockIdx.x * blockDim.x + threadIdx.x;
    if (r >= BS) return;
    const float* zr = z1 + (size_t)r * HC;
    float z2[32];
#pragma unroll 4
    for (int j = 0; j < 32; ++j) {
        float s = b2[j];
        const float* w = W2 + j * 64;
        for (int k = 0; k < 64; ++k) s += zr[k] * w[k];
        z2[j] = fmaxf(s, 0.0f);
    }
    float z3[16];
#pragma unroll 4
    for (int j = 0; j < 16; ++j) {
        float s = b3[j];
        const float* w = W3 + j * 32;
        for (int k = 0; k < 32; ++k) s += z2[k] * w[k];
        z3[j] = fmaxf(s, 0.0f);
    }
    float lg[3];
#pragma unroll
    for (int j = 0; j < 3; ++j) {
        float s = b4[j];
        const float* w = W4 + j * 16;
        for (int k = 0; k < 16; ++k) s += z3[k] * w[k];
        lg[j] = s;
    }
    float mx = fmaxf(lg[0], fmaxf(lg[1], lg[2]));
    float e0 = __expf(lg[0] - mx), e1 = __expf(lg[1] - mx), e2 = __expf(lg[2] - mx);
    float inv = 1.0f / (e0 + e1 + e2);
    out[r * 3 + 0] = e0 * inv;
    out[r * 3 + 1] = e1 * inv;
    out[r * 3 + 2] = e2 * inv;
}

extern "C" void kernel_launch(void* const* d_in, const int* in_sizes, int n_in,
                              void* d_out, int out_size, void* d_ws, size_t ws_size,
                              hipStream_t stream) {
    (void)in_sizes; (void)n_in; (void)out_size; (void)ws_size;
    const float* x   = (const float*)d_in[0];
    const int*   ei  = (const int*)d_in[1];
    const float* ea  = (const float*)d_in[2];
    const float* Wg  = (const float*)d_in[3];
    const float* wih = (const float*)d_in[4];
    const float* whh = (const float*)d_in[5];
    const float* bih = (const float*)d_in[6];
    const float* bhh = (const float*)d_in[7];
    const float* W1  = (const float*)d_in[8];
    const float* b1  = (const float*)d_in[9];
    const float* W2  = (const float*)d_in[10];
    const float* b2  = (const float*)d_in[11];
    const float* W3  = (const float*)d_in[12];
    const float* b3  = (const float*)d_in[13];
    const float* W4  = (const float*)d_in[14];
    const float* b4  = (const float*)d_in[15];
    float* out = (float*)d_out;

    const size_t NB = (size_t)NN * HC * sizeof(float);   // 20.97 MB
    char* ws = (char*)d_ws;
    float* h    = (float*)(ws);
    float* m    = (float*)(ws + NB);
    float* aggr = (float*)(ws + 2 * NB);
    float* z1   = (float*)(ws + 3 * NB);                 // 512*64 floats

    pad_kernel<<<(NN * HC) / 256, 256, 0, stream>>>(x, h);

    for (int layer = 0; layer < 2; ++layer) {
        gemm_m_kernel<<<((NN / 16) * 4) / 8, 256, 0, stream>>>(h, Wg + layer * HC * HC, m);
        hipMemsetAsync(aggr, 0, NB, stream);
        scatter_kernel<<<(EE * 16) / 256, 256, 0, stream>>>(ei, ea, m, aggr);
        gru_kernel<<<NN / 16, 128, 0, stream>>>(h, aggr, wih, whh, bih, bhh);
    }

    hipMemsetAsync(z1, 0, (size_t)BS * HC * sizeof(float), stream);
    head1_kernel<<<(32 * 4 * SPLITK) / 8, 256, 0, stream>>>(h, W1, z1);
    bias_relu_kernel<<<(BS * HC) / 256, 256, 0, stream>>>(z1, b1);
    tail_kernel<<<2, 256, 0, stream>>>(z1, W2, b2, W3, b3, W4, b4, out);
}
